// SelfAttention_46127948759290
// MI455X (gfx1250) — matrix operand order repaired and checked by reading, not compiled, not run on hardware
//
#include <hip/hip_runtime.h>
#include <hip/hip_bf16.h>
#include <hip/hip_fp16.h>

// ---------------------------------------------------------------------------
// Self-attention (flash-attention style) + output projection for gfx1250.
//   N=2, S=2048, E=1024, H=16, D=64.
//   energy = q.k^T (per n,h), softmax((mask? e : -1e20)/32), out = attn.v,
//   final = out @ W^T + b.
// Structure:
//   0) one-shot pre-pass: Q (pre-scaled by 1/sqrt(E)), K, W -> f16; V -> f16
//      AND transposed to [n,h,d,s] so LDS tiles are B-operand friendly,
//   1) flash attention: 4 waves/WG share double-buffered f16 K/V tiles moved
//      by GLOBAL_LOAD_ASYNC_TO_LDS_B128 (+ s_wait_asynccnt); all fragment
//      builds are contiguous ds_load_b128; math on v_wmma_f32_16x16x32_f16,
//   2) output projection GEMM on WMMA, f32 accumulate, bias add.
// ---------------------------------------------------------------------------

#define EMBED  1024
#define HEADS  16
#define HDIM   64
#define SEQ    2048
#define NBATCH 2

typedef __attribute__((ext_vector_type(16))) _Float16 v16h;
typedef __attribute__((ext_vector_type(8)))  float    v8f;

// masked energy, pre-scaled: -1e20 / sqrt(1024)
#define MASKED_E (-3.125e18f)

// A-matrix (16x32 f16) half-index -> K mapping (ISA 7.12.2):
//   lane l: row M = l&15, group g = l>>4;  half j -> K = (j&7) + 8g + 16(j>>3)
__device__ __forceinline__ int amap_k(int j, int g) {
    return (j & 7) + 8 * g + 16 * (j >> 3);
}

__device__ __forceinline__ v8f wmma_f16(v16h a, v16h b, v8f c) {
    // (neg_a, A, neg_b, B, c_mod, C, reuse_a, reuse_b)
    return __builtin_amdgcn_wmma_f32_16x16x32_f16(false, a, false, b,
                                                  (short)0, c, false, false);
}

// Async DMA: one b128 (16B) per active lane, global -> LDS. ASYNCcnt++.
__device__ __forceinline__ void async_ld_b128(uint32_t lds_off, const void* gptr) {
    asm volatile("global_load_async_to_lds_b128 %0, %1, off"
                 :: "v"(lds_off), "v"(gptr) : "memory");
}

// ---------------------------------------------------------------------------
// Kernel 0a: elementwise (scale * f32) -> f16. One float4 per thread.
// ---------------------------------------------------------------------------
__global__ __launch_bounds__(256)
void cvt_f32_to_f16_kernel(const float* __restrict__ src,
                           _Float16*    __restrict__ dst,
                           float scale)
{
    const size_t i = (size_t)blockIdx.x * blockDim.x + threadIdx.x;
    float4 w = ((const float4*)src)[i];
    dst[4 * i + 0] = (_Float16)(w.x * scale);
    dst[4 * i + 1] = (_Float16)(w.y * scale);
    dst[4 * i + 2] = (_Float16)(w.z * scale);
    dst[4 * i + 3] = (_Float16)(w.w * scale);
}

// ---------------------------------------------------------------------------
// Kernel 0b: V [N,S,E] f32 -> Vt [N,H,D,S] f16 (per-head transpose).
// 32x32 tiles via LDS; both global sides coalesced.
// ---------------------------------------------------------------------------
__global__ __launch_bounds__(256)
void v_transpose_kernel(const float* __restrict__ V, _Float16* __restrict__ Vt)
{
    __shared__ _Float16 t[32][33];           // +1 pad against bank conflicts

    const int s0 = blockIdx.x * 32;          // seq tile
    const int d0 = blockIdx.y * 32;          // dim tile within head
    const int nh = blockIdx.z;               // n*HEADS + h
    const int n  = nh / HEADS;
    const int h  = nh % HEADS;

    const int r  = threadIdx.x >> 3;         // 0..31
    const int c4 = (threadIdx.x & 7) * 4;    // 0,4,...,28

    float4 v = *(const float4*)(V + (size_t)(n * SEQ + s0 + r) * EMBED
                                  + h * HDIM + d0 + c4);
    t[r][c4 + 0] = (_Float16)v.x;
    t[r][c4 + 1] = (_Float16)v.y;
    t[r][c4 + 2] = (_Float16)v.z;
    t[r][c4 + 3] = (_Float16)v.w;
    __syncthreads();

    // write transposed: row = dim, cols = 4 consecutive seq positions
    _Float16* orow = Vt + ((size_t)nh * HDIM + d0 + r) * SEQ + s0 + c4;
    orow[0] = t[c4 + 0][r];
    orow[1] = t[c4 + 1][r];
    orow[2] = t[c4 + 2][r];
    orow[3] = t[c4 + 3][r];
}

// ---------------------------------------------------------------------------
// Kernel 1: fused attention. 4 waves/WG; wave w owns a 16-query tile; the WG
// shares double-buffered async-staged f16 tiles:
//   K block: [key(32)][dim(64)]   Vt block: [dim(64)][key(32)]
// Writes softmax(QK^T/32) @ V as f16 into X, laid out [N*S, E].
// ---------------------------------------------------------------------------
__global__ __launch_bounds__(128)
void attn_flash_kernel(const _Float16* __restrict__ Vt,
                       const _Float16* __restrict__ Kh,
                       const _Float16* __restrict__ Qh,
                       const int*      __restrict__ Mask,
                       _Float16*       __restrict__ X)
{
    __shared__ _Float16 lds_k[2][32 * 64];   // [buf][key][dim]  (4KB each)
    __shared__ _Float16 lds_v[2][64 * 32];   // [buf][dim][key]  (4KB each)
    __shared__ _Float16 lds_p[4][16 * 32];   // per-wave P tile  (1KB each)

    const int tid  = threadIdx.x;        // 0..127
    const int lane = tid & 31;
    const int wave = tid >> 5;           // 0..3
    const int col  = lane & 15;          // N index inside a 16-wide tile
    const int g    = lane >> 4;          // half-wave group

    const int q0 = blockIdx.x * 64 + wave * 16;   // this wave's query tile
    const int h  = blockIdx.y;
    const int n  = blockIdx.z;

    const _Float16* vtb = Vt + (size_t)(n * HEADS + h) * HDIM * SEQ;

    // ---- async stage of one 32-key block (4KB K + 4KB Vt) into buffer b ----
    // 256 x 16B chunks per matrix; 128 threads x 2 chunks; 4 async ops/wave.
    auto issue_block = [&](int k0, int b) {
#pragma unroll
        for (int i = 0; i < 2; ++i) {
            const int c = tid * 2 + i;            // chunk 0..255
            const uint32_t lo = (uint32_t)(c * 16);
            // K: row-major [key][dim]; 8 chunks per 128B row
            const int kkey = c >> 3, kci = c & 7;
            async_ld_b128((uint32_t)(uintptr_t)lds_k[b] + lo,
                          Kh + (size_t)(n * SEQ + k0 + kkey) * EMBED
                             + h * HDIM + kci * 8);
            // Vt: dim-major [dim][key]; 4 chunks per 64B row
            const int vdim = c >> 2, vci = c & 3;
            async_ld_b128((uint32_t)(uintptr_t)lds_v[b] + lo,
                          vtb + (size_t)vdim * SEQ + k0 + vci * 8);
        }
    };

    // ---- Q tile as two 16x32 f16 A-fragments (K-dims 0..31 / 32..63) ----
    const _Float16* qrow = Qh + (size_t)(n * SEQ + q0 + col) * EMBED + h * HDIM;
    v16h qa0, qa1;
#pragma unroll
    for (int jj = 0; jj < 8; ++jj) {
        const int pb = 2 * jj + 16 * g; // K of even half
        qa0[2 * jj]     = qrow[pb];
        qa0[2 * jj + 1] = qrow[pb + 1];
        qa1[2 * jj]     = qrow[32 + pb];
        qa1[2 * jj + 1] = qrow[32 + pb + 1];
    }

    // ---- online-softmax state: rows r+8g live in accumulator slot r ----
    v8f o[4] = {{}, {}, {}, {}};         // O tile: 16 q x 64 dims (4 N-tiles)
    float mrow[8], lrow[8];
#pragma unroll
    for (int r = 0; r < 8; ++r) { mrow[r] = -3.0e38f; lrow[r] = 0.0f; }

    issue_block(0, 0);                   // prologue: fill buffer 0

    for (int kb = 0; kb < SEQ / 32; ++kb) {
        const int k0 = kb * 32;
        const int b  = kb & 1;

        if (kb + 1 < SEQ / 32) {
            issue_block(k0 + 32, b ^ 1);                     // overlap next
            asm volatile("s_wait_asynccnt 4" ::: "memory");  // prev group done
        } else {
            asm volatile("s_wait_asynccnt 0" ::: "memory");
        }
        __syncthreads();   // all waves' DMA for buffer b complete

        // ---- scores (Q pre-scaled): two 16x16 tiles ----
        v8f s[2];
#pragma unroll
        for (int t = 0; t < 2; ++t) {
            v16h b0, b1;   // B-frag: lane=col N (key), half j -> K-dim j+16g
#pragma unroll
            for (int j = 0; j < 16; ++j) {
                const int key = t * 16 + col;
                b0[j] = lds_k[b][key * 64 + (j + 16 * g)];
                b1[j] = lds_k[b][key * 64 + 32 + (j + 16 * g)];
            }
            v8f acc = {};
            acc  = wmma_f16(qa0, b0, acc);
            s[t] = wmma_f16(qa1, b1, acc);
        }

        // ---- mask (C layout: slot r is q-row r+8g, lane col = key) ----
#pragma unroll
        for (int t = 0; t < 2; ++t) {
#pragma unroll
            for (int r = 0; r < 8; ++r) {
                const int qi = q0 + r + 8 * g;
                const int ki = k0 + t * 16 + col;
                const int mv = Mask[(size_t)n * SEQ * SEQ + (size_t)qi * SEQ + ki];
                s[t][r] = (mv == 0) ? MASKED_E : s[t][r];
            }
        }

        // ---- online softmax update ----
        float alpha[8];
#pragma unroll
        for (int r = 0; r < 8; ++r) {
            float mx = fmaxf(s[0][r], s[1][r]);
#pragma unroll
            for (int w = 1; w < 16; w <<= 1)
                mx = fmaxf(mx, __shfl_xor(mx, w, 32));
            const float mn = fmaxf(mrow[r], mx);
            const float p0 = __expf(s[0][r] - mn);
            const float p1 = __expf(s[1][r] - mn);
            float rs = p0 + p1;
#pragma unroll
            for (int w = 1; w < 16; w <<= 1)
                rs += __shfl_xor(rs, w, 32);
            const float a = __expf(mrow[r] - mn);
            lrow[r]  = lrow[r] * a + rs;
            mrow[r]  = mn;
            alpha[r] = a;
            // stash probabilities for layout conversion (C -> A frag)
            lds_p[wave][(r + 8 * g) * 32 + col]      = (_Float16)p0;
            lds_p[wave][(r + 8 * g) * 32 + 16 + col] = (_Float16)p1;
        }
#pragma unroll
        for (int r = 0; r < 8; ++r) {
            o[0][r] *= alpha[r]; o[1][r] *= alpha[r];
            o[2][r] *= alpha[r]; o[3][r] *= alpha[r];
        }

        // ---- P as 16x32 A-fragment (same-wave LDS, in-order) ----
        v16h pa;
#pragma unroll
        for (int j = 0; j < 16; ++j)
            pa[j] = lds_p[wave][col * 32 + (j + 16 * g)];

        // ---- O += P @ V_block: Vt tile is [dim][key], contiguous reads ----
#pragma unroll
        for (int t = 0; t < 4; ++t) {
            v16h vb;   // lane=col (dim t*16+col), half j -> key j+16g
#pragma unroll
            for (int j = 0; j < 16; ++j)
                vb[j] = lds_v[b][(t * 16 + col) * 32 + (j + 16 * g)];
            o[t] = wmma_f16(pa, vb, o[t]);
        }
        __syncthreads();   // all waves done reading buffer b before reuse
    }

    // ---- normalize by row sums, emit f16 into X[N*S, E] ----
#pragma unroll
    for (int t = 0; t < 4; ++t) {
#pragma unroll
        for (int r = 0; r < 8; ++r) {
            const float val = o[t][r] / lrow[r];
            X[(size_t)(n * SEQ + q0 + r + 8 * g) * EMBED
              + h * HDIM + t * 16 + col] = (_Float16)val;
        }
    }
}

// ---------------------------------------------------------------------------
// Kernel 2: Out = X @ Wh^T + bias.  One wave32 per 16x64 output tile.
//   M = N*S = 4096, K = 1024, Nc = 1024. X/Wh are f16 (ws), Out is f32.
// ---------------------------------------------------------------------------
__global__ __launch_bounds__(32)
void out_proj_kernel(const _Float16* __restrict__ X,
                     const _Float16* __restrict__ Wh,
                     const float*    __restrict__ Bias,
                     float*          __restrict__ Out)
{
    const int lane = threadIdx.x;
    const int col  = lane & 15;
    const int g    = lane >> 4;

    const int m0 = blockIdx.x * 16;    // output row tile
    const int n0 = blockIdx.y * 64;    // output col tile (4 x 16)

    v8f c[4] = {{}, {}, {}, {}};
    const _Float16* xrow = X + (size_t)(m0 + col) * EMBED;

    for (int kb = 0; kb < EMBED; kb += 32) {
        // A-frag from f16 activations
        v16h a;
#pragma unroll
        for (int jj = 0; jj < 8; ++jj) {
            const int kk = kb + 2 * jj + 16 * g;
            a[2 * jj]     = xrow[kk];
            a[2 * jj + 1] = xrow[kk + 1];
        }
        // B-frags: B[k][nn] = Wh[nn][k]  (lane=output col, half j -> k=j+16g)
#pragma unroll
        for (int t = 0; t < 4; ++t) {
            const _Float16* wrow =
                Wh + (size_t)(n0 + t * 16 + col) * EMBED + kb + 16 * g;
            v16h bb;
#pragma unroll
            for (int j = 0; j < 16; ++j)
                bb[j] = wrow[j];
            c[t] = wmma_f16(a, bb, c[t]);
        }
    }

#pragma unroll
    for (int t = 0; t < 4; ++t) {
        const float bias = Bias[n0 + t * 16 + col];
#pragma unroll
        for (int r = 0; r < 8; ++r) {
            Out[(size_t)(m0 + r + 8 * g) * EMBED + n0 + t * 16 + col] =
                c[t][r] + bias;
        }
    }
}

// ---------------------------------------------------------------------------
// Launch. Input order (setup_inputs): values, keys, query, mask, W_out, b_out.
// d_ws layout (f16 elements):
//   [ 0M,  4M)  Qh   [N*S, E]   (pre-scaled by 1/sqrt(E))
//   [ 4M,  8M)  Kh   [N*S, E]
//   [ 8M, 12M)  Vt   [N,H,D,S]  (per-head transposed V)
//   [12M, 16M)  X    [N*S, E]   (attention output)
//   [16M, 17M)  Wh   [E, E]
// Total 34 MB of workspace.
// ---------------------------------------------------------------------------
extern "C" void kernel_launch(void* const* d_in, const int* in_sizes, int n_in,
                              void* d_out, int out_size, void* d_ws, size_t ws_size,
                              hipStream_t stream)
{
    const float* values = (const float*)d_in[0];
    const float* keys   = (const float*)d_in[1];
    const float* query  = (const float*)d_in[2];
    const int*   mask   = (const int*)  d_in[3];
    const float* W_out  = (const float*)d_in[4];
    const float* b_out  = (const float*)d_in[5];
    float*       out    = (float*)d_out;

    const size_t TOK = (size_t)NBATCH * SEQ * EMBED;   // 4M elements
    _Float16* Qh = (_Float16*)d_ws;
    _Float16* Kh = Qh + TOK;
    _Float16* Vt = Kh + TOK;
    _Float16* X  = Vt + TOK;
    _Float16* Wh = X + TOK;

    // one-shot pre-pass: conversions (+ Q scaling) and per-head V transpose
    cvt_f32_to_f16_kernel<<<(int)(TOK / 4 / 256), 256, 0, stream>>>(
        query, Qh, 0.03125f);                        // fold 1/sqrt(E) into Q
    cvt_f32_to_f16_kernel<<<(int)(TOK / 4 / 256), 256, 0, stream>>>(
        keys, Kh, 1.0f);
    cvt_f32_to_f16_kernel<<<(EMBED * EMBED / 4) / 256, 256, 0, stream>>>(
        W_out, Wh, 1.0f);
    dim3 gt(SEQ / 32, HDIM / 32, NBATCH * HEADS);    // 64 x 2 x 32
    v_transpose_kernel<<<gt, 256, 0, stream>>>(values, Vt);

    dim3 ga(SEQ / 64, HEADS, NBATCH);          // 32 x 16 x 2 WGs of 4 waves
    attn_flash_kernel<<<ga, 128, 0, stream>>>(Vt, Kh, Qh, mask, X);

    dim3 gp((NBATCH * SEQ) / 16, EMBED / 64);  // 256 x 16 waves
    out_proj_kernel<<<gp, 32, 0, stream>>>(X, Wh, b_out, out);
}
